// LSTMLayer_58308476011097
// MI455X (gfx1250) — matrix operand order repaired and checked by reading, not compiled, not run on hardware
//
#include <hip/hip_runtime.h>
#include <hip/hip_bf16.h>

#define NWG   32
#define TLEN  2048
#define BATCH 32
#define DDIM  512
#define HDIM  512
#define WSTR  520   // padded LDS row stride (elements) to avoid bank aliasing

typedef __attribute__((ext_vector_type(16))) __bf16 v16bf;
typedef __attribute__((ext_vector_type(8)))  float  v8f;
typedef int v4i __attribute__((vector_size(16)));   // int4 vector, matches builtin pointee

union FragBF {
    v16bf v;
    uint4 q[2];
};

#if __has_builtin(__builtin_amdgcn_global_load_async_to_lds_b128)
#define HAS_ASYNC_LDS 1
#define AS1 __attribute__((address_space(1)))
#define AS3 __attribute__((address_space(3)))
#else
#define HAS_ASYNC_LDS 0
#endif

__global__ void sync_init_kernel(unsigned* s) {
    s[0] = 0u;
    s[1] = 0u;
}

__launch_bounds__(256, 1)
__global__ void lstm_persistent_kernel(const float* __restrict__ x,     // [32,2048,512]
                                       const float* __restrict__ h0,    // [32,512]
                                       const float* __restrict__ Wih,   // [2048,512] gate-major (i,f,g,o)
                                       const float* __restrict__ Whh,   // [2048,512]
                                       const float* __restrict__ bias,  // [2048]
                                       float* __restrict__ out,         // [32,512]
                                       unsigned* __restrict__ sync,     // [2] {arrive, generation}
                                       __bf16* __restrict__ hbuf0,      // [32,512] bf16
                                       __bf16* __restrict__ hbuf1)      // [32,512] bf16
{
    // LDS: stationary bf16 weight slices + per-step activation tiles
    __shared__ __align__(16) __bf16 Wih_s[64 * WSTR];   // rows: q*16+j  -> W_ih[q*512 + wg*16 + j][:]
    __shared__ __align__(16) __bf16 Whh_s[64 * WSTR];
    __shared__ __align__(16) __bf16 xs[BATCH * WSTR];   // x_t  bf16
    __shared__ __align__(16) __bf16 hs[BATCH * WSTR];   // h_t  bf16
    __shared__ __align__(16) float  gbuf[BATCH * 64];   // activated gates [b][q*16+j]

    const int tid  = threadIdx.x;
    const int wg   = blockIdx.x;
    const int wave = tid >> 5;       // 0..7
    const int lane = tid & 31;
    const int mt   = wave & 1;       // M tile (batch rows 0-15 / 16-31)
    const int qg   = wave >> 1;      // gate index 0..3 (i,f,g,o)
    const int ln   = lane & 15;
    const int lhi  = lane >> 4;      // 0/1: lane half

    // ---- Phase 0: stage bf16 weight slices into LDS (stationary for all T) ----
    for (int lr4 = tid; lr4 < 64 * 4; lr4 += 256) {       // 4 threads per row, 128 cols each
        const int lr   = lr4 >> 2;
        const int cseg = (lr4 & 3) * 128;
        const int grow = ((lr >> 4) << 9) + (wg << 4) + (lr & 15);   // q*512 + wg*16 + j
        const float* gi = Wih + (size_t)grow * DDIM + cseg;
        const float* gh = Whh + (size_t)grow * HDIM + cseg;
        __bf16* di = Wih_s + lr * WSTR + cseg;
        __bf16* dh = Whh_s + lr * WSTR + cseg;
        for (int c = 0; c < 128; c += 4) {
            float4 a = *(const float4*)(gi + c);
            di[c + 0] = (__bf16)a.x; di[c + 1] = (__bf16)a.y;
            di[c + 2] = (__bf16)a.z; di[c + 3] = (__bf16)a.w;
            float4 b4 = *(const float4*)(gh + c);
            dh[c + 0] = (__bf16)b4.x; dh[c + 1] = (__bf16)b4.y;
            dh[c + 2] = (__bf16)b4.z; dh[c + 3] = (__bf16)b4.w;
        }
    }

    // ---- Phase 0b: publish h0 slice (bf16) for this WG's 16 columns ----
    const int ub = tid >> 3;           // batch row 0..31
    const int uj = (tid & 7) * 2;      // 2 h-dims per thread
    const int ucol = wg * 16 + uj;
    {
        float a0 = h0[(size_t)ub * HDIM + ucol];
        float a1 = h0[(size_t)ub * HDIM + ucol + 1];
        hbuf0[(size_t)ub * HDIM + ucol]     = (__bf16)a0;
        hbuf0[(size_t)ub * HDIM + ucol + 1] = (__bf16)a1;
    }

    // per-thread cell state (c_prev = 0)
    float c0 = 0.f, c1 = 0.f;
    // per-lane bias for this wave's output column
    const float biasv = bias[qg * HDIM + wg * 16 + ln];
    // branchless activation constants (wave-uniform):
    //   sigmoid: a = 1*sig(1*v) + 0 ; tanh: a = 2*sig(2*v) - 1
    const float act_si = (qg == 2) ? 2.f : 1.f;
    const float act_so = (qg == 2) ? 2.f : 1.f;
    const float act_b  = (qg == 2) ? -1.f : 0.f;

    // ---- grid barrier state ----
    int bar = 0;
    auto grid_barrier = [&](void) {
        ++bar;
        __syncthreads();
        __builtin_amdgcn_fence(__ATOMIC_RELEASE, "agent");
        if (tid == 0) {
            unsigned prev = atomicAdd(&sync[0], 1u);
            if (prev == (unsigned)(NWG * bar - 1)) {
                atomicAdd(&sync[1], 1u);
            }
            while (__hip_atomic_load(&sync[1], __ATOMIC_RELAXED,
                                     __HIP_MEMORY_SCOPE_AGENT) < (unsigned)bar) {
                __builtin_amdgcn_s_sleep(1);
            }
        }
        __syncthreads();
        __builtin_amdgcn_fence(__ATOMIC_ACQUIRE, "agent");
    };

    grid_barrier();   // h0 visible everywhere

    // fragment base offsets (element units)
    const int arow  = (mt * 16 + ln) * WSTR;   // A: row = batch
    const int koffA = lhi * 8;                 // +8 elems for high lane half
    const int brow  = (qg * 16 + ln) * WSTR;   // B: row = gate column (W row)
    const int koffB = lhi * 8;                 // +8 elems for high lane half

    const int d0 = (tid & 7) * 64;             // per-thread 64-element column chunk

    for (int t = 0; t < TLEN; ++t) {
        const __bf16* hcur  = (t & 1) ? hbuf1 : hbuf0;
        __bf16*       hnext = (t & 1) ? hbuf0 : hbuf1;

        // ---- stage x_t (f32 -> bf16) and h_t (bf16 copy) into LDS ----
        {
            const float* xp = x + ((size_t)ub * TLEN + t) * DDIM + d0;
            __bf16* xd = xs + ub * WSTR + d0;
            for (int c = 0; c < 64; c += 4) {
                float4 a = *(const float4*)(xp + c);
                xd[c + 0] = (__bf16)a.x; xd[c + 1] = (__bf16)a.y;
                xd[c + 2] = (__bf16)a.z; xd[c + 3] = (__bf16)a.w;
            }
#if HAS_ASYNC_LDS
            // async copy h_t directly into LDS (ASYNCcnt path); INST_OFFSET is
            // applied to both the global and LDS address per the ISA.
            AS1 v4i* gsrc = (AS1 v4i*)(void*)
                (const_cast<__bf16*>(hcur) + (size_t)ub * HDIM + d0);
            AS3 v4i* ldst = (AS3 v4i*)(void*)(hs + ub * WSTR + d0);
            __builtin_amdgcn_global_load_async_to_lds_b128(gsrc, ldst, 0,   0);
            __builtin_amdgcn_global_load_async_to_lds_b128(gsrc, ldst, 16,  0);
            __builtin_amdgcn_global_load_async_to_lds_b128(gsrc, ldst, 32,  0);
            __builtin_amdgcn_global_load_async_to_lds_b128(gsrc, ldst, 48,  0);
            __builtin_amdgcn_global_load_async_to_lds_b128(gsrc, ldst, 64,  0);
            __builtin_amdgcn_global_load_async_to_lds_b128(gsrc, ldst, 80,  0);
            __builtin_amdgcn_global_load_async_to_lds_b128(gsrc, ldst, 96,  0);
            __builtin_amdgcn_global_load_async_to_lds_b128(gsrc, ldst, 112, 0);
#if __has_builtin(__builtin_amdgcn_s_wait_asynccnt)
            __builtin_amdgcn_s_wait_asynccnt(0);
#else
            asm volatile("s_wait_asynccnt 0" ::: "memory");
#endif
#else
            const uint4* hsrc = (const uint4*)(hcur + (size_t)ub * HDIM + d0);
            uint4* hdst = (uint4*)(hs + ub * WSTR + d0);
            for (int c = 0; c < 8; ++c) hdst[c] = hsrc[c];
#endif
        }
        __syncthreads();

        // prefetch next timestep's x chunk into the cache hierarchy
        if (t + 1 < TLEN) {
            __builtin_prefetch(x + ((size_t)ub * TLEN + (t + 1)) * DDIM + d0, 0, 3);
        }

        // ---- WMMA: gates tile = x_t @ W_ih^T + h_t @ W_hh^T + bias ----
        v8f accx, acch;
        #pragma unroll
        for (int i = 0; i < 8; ++i) { accx[i] = biasv; acch[i] = 0.f; }

        #pragma unroll 4
        for (int ks = 0; ks < 16; ++ks) {
            const int kb = ks * 32;
            FragBF ax, ah, bi, bh;
            ax.q[0] = *(const uint4*)(xs + arow + kb + koffA);
            ax.q[1] = *(const uint4*)(xs + arow + kb + 16 + koffA);
            ah.q[0] = *(const uint4*)(hs + arow + kb + koffA);
            ah.q[1] = *(const uint4*)(hs + arow + kb + 16 + koffA);
            bi.q[0] = *(const uint4*)(Wih_s + brow + kb + koffB);
            bi.q[1] = *(const uint4*)(Wih_s + brow + kb + koffB + 16);
            bh.q[0] = *(const uint4*)(Whh_s + brow + kb + koffB);
            bh.q[1] = *(const uint4*)(Whh_s + brow + kb + koffB + 16);
            accx = __builtin_amdgcn_wmma_f32_16x16x32_bf16(
                       false, ax.v, false, bi.v, (short)0, accx, false, false);
            acch = __builtin_amdgcn_wmma_f32_16x16x32_bf16(
                       false, ah.v, false, bh.v, (short)0, acch, false, false);
        }

        // ---- branchless activations (sigmoid / tanh via scaled sigmoid) ----
        #pragma unroll
        for (int r = 0; r < 8; ++r) {
            float v  = accx[r] + acch[r];
            float sg = 1.f / (1.f + __expf(-act_si * v));
            float a  = __builtin_fmaf(act_so, sg, act_b);
            gbuf[(mt * 16 + lhi * 8 + r) * 64 + qg * 16 + ln] = a;
        }
        __syncthreads();

        // ---- elementwise cell update: 2 (b,j) elements per thread ----
        {
            const float* gr = gbuf + ub * 64 + uj;
            float i0 = gr[0],  i1 = gr[1];
            float f0 = gr[16], f1 = gr[17];
            float g0 = gr[32], g1 = gr[33];
            float o0 = gr[48], o1 = gr[49];
            c0 = f0 * c0 + i0 * g0;
            c1 = f1 * c1 + i1 * g1;
            float e0 = __expf(-2.f * c0), th0 = (1.f - e0) / (1.f + e0);
            float e1 = __expf(-2.f * c1), th1 = (1.f - e1) / (1.f + e1);
            float hv0 = o0 * th0, hv1 = o1 * th1;
            hnext[(size_t)ub * HDIM + ucol]     = (__bf16)hv0;
            hnext[(size_t)ub * HDIM + ucol + 1] = (__bf16)hv1;
            if (t == TLEN - 1) {
                out[(size_t)ub * HDIM + ucol]     = hv0;
                out[(size_t)ub * HDIM + ucol + 1] = hv1;
            }
        }

        grid_barrier();   // h_{t+1} published; safe to reuse buffers
    }
}

extern "C" void kernel_launch(void* const* d_in, const int* in_sizes, int n_in,
                              void* d_out, int out_size, void* d_ws, size_t ws_size,
                              hipStream_t stream) {
    const float* x    = (const float*)d_in[0];   // [32,2048,512]
    const float* h0   = (const float*)d_in[1];   // [32,512]
    const float* Wih  = (const float*)d_in[2];   // [2048,512]
    const float* Whh  = (const float*)d_in[3];   // [2048,512]
    const float* bias = (const float*)d_in[4];   // [2048]
    float* out = (float*)d_out;

    unsigned char* ws = (unsigned char*)d_ws;
    unsigned* sync  = (unsigned*)ws;                              // 2 counters
    __bf16*   hbuf0 = (__bf16*)(ws + 256);                        // 32 KB
    __bf16*   hbuf1 = (__bf16*)(ws + 256 + BATCH * HDIM * 2);     // 32 KB

    sync_init_kernel<<<1, 1, 0, stream>>>(sync);
    lstm_persistent_kernel<<<NWG, 256, 0, stream>>>(
        x, h0, Wih, Whh, bias, out, sync, hbuf0, hbuf1);
}
